// KNN_Block_4243427688838
// MI455X (gfx1250) — matrix operand order repaired and checked by reading, not compiled, not run on hardware
//
#include <hip/hip_runtime.h>

typedef __attribute__((ext_vector_type(16))) __bf16       v16bf;
typedef __attribute__((ext_vector_type(8)))  float        v8f;
typedef __attribute__((ext_vector_type(4)))  unsigned int u32x4;

#define B_  16
#define C_  128
#define N_  2000
#define K_  9
#define NT_ (N_ / 16)      // 125 row tiles
#define P1_ (B_ * N_ * 3)  // 96000 conv1 columns
#define P2_ (B_ * N_)      // 32000 conv2 columns

union Frag { v16bf v; u32x4 u[2]; };

__device__ __forceinline__ v8f wmma_bf16(Frag a, Frag b, v8f c) {
  // D = A(16x32 bf16) x B(32x16 bf16) + C(16x16 f32)
  return __builtin_amdgcn_wmma_f32_16x16x32_bf16(false, a.v, false, b.v,
                                                 (short)0, c, false, false);
}

// ---------------------------------------------------------------- prep: xT (f32 + bf16)
__global__ __launch_bounds__(256) void prep_x_kernel(const float* __restrict__ feat,
                                                     float* __restrict__ xTf,
                                                     unsigned short* __restrict__ xTh) {
  int g = blockIdx.x * 256 + threadIdx.x;
  if (g >= B_ * N_ * C_) return;
  int c = g & 127;
  int n = (g >> 7) % N_;
  int b = g / (N_ * C_);
  float v = feat[((size_t)(b * C_ + c)) * N_ + n];
  xTf[g] = v;
  ((__bf16*)xTh)[g] = (__bf16)v;
}

// ---------------------------------------------------------------- prep: xx = ||x||^2
__global__ __launch_bounds__(256) void xx_kernel(const float* __restrict__ xTf,
                                                 float* __restrict__ xx) {
  int g = blockIdx.x * 256 + threadIdx.x;
  if (g >= B_ * N_) return;
  const float* row = xTf + (size_t)g * C_;
  float s = 0.f;
  for (int c = 0; c < C_; ++c) s += row[c] * row[c];
  xx[g] = s;
}

// ---------------------------------------------------------------- prep: weights -> bf16, K-major
__global__ __launch_bounds__(256) void wprep_kernel(const float* __restrict__ w1f,
                                                    const float* __restrict__ w2f,
                                                    unsigned short* __restrict__ w1h,
                                                    unsigned short* __restrict__ w2h) {
  int g = blockIdx.x * 256 + threadIdx.x;
  if (g < C_ * 768) {                 // w1h[o][k], k = r*256 + i  <-  w1[o][i][0][r]
    int o = g / 768, k = g % 768;
    int r = k >> 8, i = k & 255;
    ((__bf16*)w1h)[g] = (__bf16)w1f[((size_t)(o * 256 + i)) * 3 + r];
  } else if (g < C_ * 768 + C_ * 384) { // w2h[o][k], k = w*128 + c  <-  w2[o][c][0][w]
    int g2 = g - C_ * 768;
    int o = g2 / 384, k = g2 % 384;
    int w = k >> 7, c = k & 127;
    ((__bf16*)w2h)[g2] = (__bf16)w2f[((size_t)(o * 128 + c)) * 3 + w];
  }
}

__global__ void zero_stats_kernel(float* st) { st[threadIdx.x] = 0.f; }

// ---------------------------------------------------------------- KNN: WMMA Gram + fused top-9
__global__ __launch_bounds__(32) void knn_kernel(const unsigned short* __restrict__ xTh_,
                                                 const float* __restrict__ xx,
                                                 int* __restrict__ idxo) {
  const __bf16* xTh = (const __bf16*)xTh_;
  int b  = blockIdx.x / NT_;
  int n0 = (blockIdx.x % NT_) * 16;
  int lane = threadIdx.x;

  __shared__ float etile[16][17];

  // A fragments: rows n0..n0+15, all of C=128 (4 k-steps of 32)
  Frag afrag[4];
  {
    const __bf16* abase = xTh + ((size_t)b * N_ + (n0 + (lane & 15))) * C_;
    int asel = (lane < 16) ? 0 : 8;
    #pragma unroll
    for (int ks = 0; ks < 4; ++ks) {
      afrag[ks].u[0] = *(const u32x4*)(abase + ks * 32 + asel);
      afrag[ks].u[1] = *(const u32x4*)(abase + ks * 32 + asel + 16);
    }
  }

  float bv[9]; int bi[9];
  #pragma unroll
  for (int t = 0; t < 9; ++t) { bv[t] = 3.4e38f; bi[t] = 0; }

  int bsel = (lane < 16) ? 0 : 8;
  int col  = lane & 15;

  for (int mt = 0; mt < NT_; ++mt) {
    int m0 = mt * 16;
    const __bf16* bbase = xTh + ((size_t)b * N_ + (m0 + col)) * C_;
    v8f acc = {0.f, 0.f, 0.f, 0.f, 0.f, 0.f, 0.f, 0.f};
    #pragma unroll
    for (int ks = 0; ks < 4; ++ks) {
      Frag bf;
      bf.u[0] = *(const u32x4*)(bbase + ks * 32 + bsel);
      bf.u[1] = *(const u32x4*)(bbase + ks * 32 + bsel + 16);
      acc = wmma_bf16(afrag[ks], bf, acc);
    }
    // ranking key = -2*G[n,m] - ||x_m||^2  (row-constant terms dropped)
    float xxm = xx[b * N_ + m0 + col];
    int mb = (lane >> 4) * 8;
    #pragma unroll
    for (int v2 = 0; v2 < 8; ++v2)
      etile[mb + v2][col] = -2.0f * acc[v2] - xxm;
    __syncthreads();
    if (lane < 16) {
      for (int j = 0; j < 16; ++j) {
        float v = etile[lane][j];
        if (v < bv[8]) {
          bv[8] = v; bi[8] = m0 + j;
          for (int t = 8; t > 0; --t) {
            if (bv[t] < bv[t - 1]) {
              float tv = bv[t]; bv[t] = bv[t - 1]; bv[t - 1] = tv;
              int   ti = bi[t]; bi[t] = bi[t - 1]; bi[t - 1] = ti;
            } else break;
          }
        }
      }
    }
    __syncthreads();
  }
  if (lane < 16) {
    int* o = idxo + ((size_t)b * N_ + n0 + lane) * K_;
    #pragma unroll
    for (int t = 0; t < K_; ++t) o[t] = bi[t];
  }
}

// ---------------------------------------------------------------- conv1: WMMA, M=128 K=768
__global__ __launch_bounds__(256) void conv1_kernel(const float* __restrict__ xTf,
                                                    const int* __restrict__ idxv,
                                                    const unsigned short* __restrict__ w1h_,
                                                    const float* __restrict__ b1,
                                                    float* __restrict__ o1) {
  __shared__ unsigned short Vlds[16 * 768];   // [col][k], k contiguous
  int tid = threadIdx.x;
  int p0  = blockIdx.x * 16;

  for (int j = 0; j < 48; ++j) {              // 12288 elements / 256 threads
    int e = tid + 256 * j;
    int colv = e / 768, k = e % 768;
    int r = k >> 8, i = k & 255;
    int p = p0 + colv;
    int w = p % 3, bn = p / 3;
    int b = bn / N_;
    const float* xc = xTf + (size_t)bn * C_;
    float v;
    if (i < 128) {
      v = xc[i];
    } else {
      int ci = i - 128;
      int nb = idxv[(size_t)bn * K_ + 3 * w + r];
      v = xc[ci] - xTf[((size_t)b * N_ + nb) * C_ + ci];
    }
    ((__bf16*)Vlds)[colv * 768 + k] = (__bf16)v;
  }
  __syncthreads();

  const __bf16* w1h = (const __bf16*)w1h_;
  int lane = tid & 31, otile = tid >> 5;       // 8 waves, one M-tile each
  int orow = otile * 16 + (lane & 15);
  const __bf16* wbase = w1h + (size_t)orow * 768;
  int asel = (lane < 16) ? 0 : 8;
  int col  = lane & 15;
  const unsigned short* vbase = Vlds + col * 768 + ((lane < 16) ? 0 : 8);

  v8f acc = {0.f, 0.f, 0.f, 0.f, 0.f, 0.f, 0.f, 0.f};
  #pragma unroll
  for (int ks = 0; ks < 24; ++ks) {
    int k0 = ks * 32;
    Frag a, bf;
    a.u[0]  = *(const u32x4*)(wbase + k0 + asel);
    a.u[1]  = *(const u32x4*)(wbase + k0 + asel + 16);
    bf.u[0] = *(const u32x4*)(vbase + k0);
    bf.u[1] = *(const u32x4*)(vbase + k0 + 16);
    acc = wmma_bf16(a, bf, acc);
  }
  int mb = otile * 16 + (lane >> 4) * 8;
  float* op = o1 + (size_t)(p0 + col) * C_;
  #pragma unroll
  for (int v2 = 0; v2 < 8; ++v2) op[mb + v2] = acc[v2] + b1[mb + v2];
}

// ---------------------------------------------------------------- per-channel sum / sumsq
__global__ __launch_bounds__(256) void stats_kernel(const float* __restrict__ y, int P,
                                                    float* __restrict__ s,
                                                    float* __restrict__ q) {
  int tid = threadIdx.x;
  int c = tid & 127;
  float ls = 0.f, lq = 0.f;
  for (int p = blockIdx.x * 2 + (tid >> 7); p < P; p += gridDim.x * 2) {
    float v = y[(size_t)p * C_ + c];
    ls += v; lq += v * v;
  }
  atomicAdd(&s[c], ls);
  atomicAdd(&q[c], lq);
}

__global__ void bnfin_kernel(const float* s, const float* q, const float* w,
                             const float* bias, float Pinv, float* a, float* cs) {
  int c = threadIdx.x;
  float mean = s[c] * Pinv;
  float var  = q[c] * Pinv - mean * mean;     // biased, matches jnp.var
  float inv  = rsqrtf(var + 1e-5f);
  float aa   = w[c] * inv;
  a[c]  = aa;
  cs[c] = bias[c] - mean * aa;
}

// ---------------------------------------------------------------- conv2: BN1+ReLU fused, WMMA K=384
__global__ __launch_bounds__(256) void conv2_kernel(const float* __restrict__ o1,
                                                    const unsigned short* __restrict__ w2h_,
                                                    const float* __restrict__ a1,
                                                    const float* __restrict__ c1,
                                                    const float* __restrict__ b2,
                                                    float* __restrict__ o2) {
  __shared__ unsigned short Vlds[16 * 384];
  int tid = threadIdx.x;
  int p0  = blockIdx.x * 16;

  for (int j = 0; j < 24; ++j) {              // 6144 elements / 256 threads
    int e = tid + 256 * j;
    int colv = e / 384, k = e % 384;          // k = w*128 + c, contiguous in o1
    int c = k & 127;
    float y = o1[(size_t)(p0 + colv) * 384 + k];
    float z = fmaxf(a1[c] * y + c1[c], 0.f);
    ((__bf16*)Vlds)[colv * 384 + k] = (__bf16)z;
  }
  __syncthreads();

  const __bf16* w2h = (const __bf16*)w2h_;
  int lane = tid & 31, otile = tid >> 5;
  int orow = otile * 16 + (lane & 15);
  const __bf16* wbase = w2h + (size_t)orow * 384;
  int asel = (lane < 16) ? 0 : 8;
  int col  = lane & 15;
  const unsigned short* vbase = Vlds + col * 384 + ((lane < 16) ? 0 : 8);

  v8f acc = {0.f, 0.f, 0.f, 0.f, 0.f, 0.f, 0.f, 0.f};
  #pragma unroll
  for (int ks = 0; ks < 12; ++ks) {
    int k0 = ks * 32;
    Frag a, bf;
    a.u[0]  = *(const u32x4*)(wbase + k0 + asel);
    a.u[1]  = *(const u32x4*)(wbase + k0 + asel + 16);
    bf.u[0] = *(const u32x4*)(vbase + k0);
    bf.u[1] = *(const u32x4*)(vbase + k0 + 16);
    acc = wmma_bf16(a, bf, acc);
  }
  int mb = otile * 16 + (lane >> 4) * 8;
  float* op = o2 + (size_t)(p0 + col) * C_;
  #pragma unroll
  for (int v2 = 0; v2 < 8; ++v2) op[mb + v2] = acc[v2] + b2[mb + v2];
}

// ---------------------------------------------------------------- BN2+ReLU, transpose to NCHW
__global__ __launch_bounds__(256) void apply_kernel(const float* __restrict__ o2,
                                                    const float* __restrict__ a2,
                                                    const float* __restrict__ c2,
                                                    float* __restrict__ out) {
  int g = blockIdx.x * 256 + threadIdx.x;
  if (g >= B_ * C_ * N_) return;
  int n = g % N_;
  int c = (g / N_) & 127;
  int b = g / (C_ * N_);
  float v = o2[((size_t)b * N_ + n) * C_ + c];
  out[g] = fmaxf(a2[c] * v + c2[c], 0.f);
}

// ---------------------------------------------------------------- launch
extern "C" void kernel_launch(void* const* d_in, const int* in_sizes, int n_in,
                              void* d_out, int out_size, void* d_ws, size_t ws_size,
                              hipStream_t stream) {
  (void)in_sizes; (void)n_in; (void)out_size; (void)ws_size;
  const float* feat = (const float*)d_in[0];
  const float* w1f  = (const float*)d_in[2];
  const float* b1   = (const float*)d_in[3];
  const float* bn1w = (const float*)d_in[4];
  const float* bn1b = (const float*)d_in[5];
  const float* w2f  = (const float*)d_in[6];
  const float* b2   = (const float*)d_in[7];
  const float* bn2w = (const float*)d_in[8];
  const float* bn2b = (const float*)d_in[9];
  float* out = (float*)d_out;

  char* ws = (char*)d_ws;
  float*          xTf  = (float*)(ws);                      // 16,384,000 B
  unsigned short* xTh  = (unsigned short*)(ws + 16384000);  //  8,192,000 B
  float*          xx   = (float*)(ws + 24576000);           //    128,000 B
  int*            idxb = (int*)(ws + 24704000);             //  1,152,000 B
  unsigned short* w1h  = (unsigned short*)(ws + 25856000);  //    196,608 B
  unsigned short* w2h  = (unsigned short*)(ws + 26052608);  //     98,304 B
  float*          o1   = (float*)(ws + 26150912);           // 49,152,000 B
  float*          o2   = (float*)(ws + 75302912);           // 16,384,000 B
  float*          st   = (float*)(ws + 91686912);           //      4,096 B
  float *s1 = st,       *q1 = st + 128, *s2 = st + 256, *q2 = st + 384;
  float *a1 = st + 512, *c1 = st + 640, *a2 = st + 768, *c2 = st + 896;

  prep_x_kernel<<<(B_ * N_ * C_ + 255) / 256, 256, 0, stream>>>(feat, xTf, xTh);
  xx_kernel<<<(B_ * N_ + 255) / 256, 256, 0, stream>>>(xTf, xx);
  wprep_kernel<<<(C_ * 768 + C_ * 384 + 255) / 256, 256, 0, stream>>>(w1f, w2f, w1h, w2h);
  zero_stats_kernel<<<1, 512, 0, stream>>>(st);

  knn_kernel<<<B_ * NT_, 32, 0, stream>>>(xTh, xx, idxb);

  conv1_kernel<<<P1_ / 16, 256, 0, stream>>>(xTf, idxb, w1h, b1, o1);
  stats_kernel<<<240, 256, 0, stream>>>(o1, P1_, s1, q1);
  bnfin_kernel<<<1, 128, 0, stream>>>(s1, q1, bn1w, bn1b, 1.0f / (float)P1_, a1, c1);

  conv2_kernel<<<P2_ / 16, 256, 0, stream>>>(o1, w2h, a1, c1, b2, o2);
  stats_kernel<<<240, 256, 0, stream>>>(o2, P2_, s2, q2);
  bnfin_kernel<<<1, 128, 0, stream>>>(s2, q2, bn2w, bn2b, 1.0f / (float)P2_, a2, c2);

  apply_kernel<<<(B_ * C_ * N_ + 255) / 256, 256, 0, stream>>>(o2, a2, c2, out);
}
